// SelfAttention_32822140076585
// MI455X (gfx1250) — matrix operand order repaired and checked by reading, not compiled, not run on hardware
//
#include <hip/hip_runtime.h>
#include <hip/hip_bf16.h>

// ---------------------------------------------------------------------------
// Problem constants (match the reference)
// ---------------------------------------------------------------------------
#define B_  8
#define C_  512
#define QK_ 64
#define L_  4096

typedef __attribute__((ext_vector_type(16))) __bf16 v16bf;
typedef __attribute__((ext_vector_type(8)))  __bf16 v8bf;
typedef __attribute__((ext_vector_type(8)))  float  v8f;

static __device__ __forceinline__ v16bf cat16(v8bf lo, v8bf hi) {
  return __builtin_shufflevector(lo, hi, 0,1,2,3,4,5,6,7,8,9,10,11,12,13,14,15);
}
// 16 contiguous bf16 (32 bytes) -> one WMMA fragment register set
static __device__ __forceinline__ v16bf ld_bf16x16(const __bf16* p) {
  v8bf lo = *(const v8bf*)(p);
  v8bf hi = *(const v8bf*)(p + 8);
  return cat16(lo, hi);
}
static __device__ __forceinline__ unsigned pack_bf16(float a, float b) {
  const unsigned u0 = (unsigned)__builtin_bit_cast(unsigned short, (__bf16)a);
  const unsigned u1 = (unsigned)__builtin_bit_cast(unsigned short, (__bf16)b);
  return u0 | (u1 << 16);
}

#define WMMA_BF16(a, b, c) \
  __builtin_amdgcn_wmma_f32_16x16x32_bf16(false, (a), false, (b), (short)0, (c), false, false)

// ---------------------------------------------------------------------------
// Kernel 0: weight prep. Stack [Wq;Wk] -> bf16 [128,512], Wv -> bf16 [512,512],
// stack [bq;bk] -> f32 [128].
// ---------------------------------------------------------------------------
__global__ __launch_bounds__(256) void prep_weights_kernel(
    const float* __restrict__ Wq, const float* __restrict__ bq,
    const float* __restrict__ Wk, const float* __restrict__ bk,
    const float* __restrict__ Wv,
    __bf16* __restrict__ wqk, __bf16* __restrict__ wv, float* __restrict__ bqk)
{
  const int i = blockIdx.x * 256 + threadIdx.x;     // grid covers 512*512
  if (i < 64 * C_)            wqk[i] = (__bf16)Wq[i];
  else if (i < 128 * C_)      wqk[i] = (__bf16)Wk[i - 64 * C_];
  if (i < C_ * C_)            wv[i]  = (__bf16)Wv[i];
  if (i < 64)                 bqk[i] = bq[i];
  else if (i < 128)           bqk[i] = bk[i - 64];
}

// ---------------------------------------------------------------------------
// Kernel 1: projection GEMM via WMMA, double-buffered LDS pipeline.
//   Y[b][m][l] = sum_c W[m][c] * x[b][c][l] + bias[m]
// Block: 8 waves, output tile 64(m) x 256(l). x tile staged to LDS as bf16,
// transposed to [l][c] with padded rows; next tile prefetched to registers
// while the current K-step computes. One barrier per K-step.
// ---------------------------------------------------------------------------
#define XSTRIDE 40   // bf16 elements per LDS row (80B: 16B aligned, low conflicts)

__global__ __launch_bounds__(256, 1) void proj_gemm_kernel(
    const __bf16* __restrict__ W, const float* __restrict__ bias,
    const float* __restrict__ x,
    __bf16* __restrict__ qb, __bf16* __restrict__ kb, __bf16* __restrict__ vb,
    int is_v)
{
  __shared__ __bf16 XT[2][256 * XSTRIDE];   // 40 KB double buffer

  const int tid  = threadIdx.x;
  const int wave = tid >> 5;
  const int lane = tid & 31;
  const int hi   = lane >> 4;
  const int ln   = lane & 15;

  const int l0 = blockIdx.x * 256;
  const int m0 = blockIdx.y * 64;
  const int b  = blockIdx.z;

  v8f acc[4][2];
#pragma unroll
  for (int mt = 0; mt < 4; ++mt) { acc[mt][0] = (v8f){}; acc[mt][1] = (v8f){}; }

  const float*  xb    = x + (size_t)b * C_ * L_ + l0 + tid;
  const __bf16* wbase = W + (size_t)(m0 + ln) * C_;

  unsigned stg[16];

  // prologue: stage K-step 0 into buffer 0
#pragma unroll
  for (int cc = 0; cc < 32; cc += 2)
    stg[cc >> 1] = pack_bf16(xb[(size_t)cc * L_], xb[(size_t)(cc + 1) * L_]);
#pragma unroll
  for (int cc = 0; cc < 32; cc += 2)
    *(unsigned*)&XT[0][tid * XSTRIDE + cc] = stg[cc >> 1];
  __syncthreads();

  for (int ks = 0; ks < C_ / 32; ++ks) {
    const int buf = ks & 1;
    const int k0  = ks * 32;

    // prefetch next x tile into registers (overlaps with WMMA below)
    if (ks < C_ / 32 - 1) {
#pragma unroll
      for (int cc = 0; cc < 32; cc += 2)
        stg[cc >> 1] = pack_bf16(xb[(size_t)(k0 + 32 + cc) * L_],
                                 xb[(size_t)(k0 + 33 + cc) * L_]);
    }

    // B fragments: lane holds column l, k-elems = c = k0 + 16*hi + e
    v16bf bf0 = ld_bf16x16(&XT[buf][(wave * 32 +      ln) * XSTRIDE + 16 * hi]);
    v16bf bf1 = ld_bf16x16(&XT[buf][(wave * 32 + 16 + ln) * XSTRIDE + 16 * hi]);

    // A fragments (batched into one clause): elems 0..15 @ c=k0+16hi+e
    v16bf af[4];
#pragma unroll
    for (int mt = 0; mt < 4; ++mt) {
      const __bf16* wr = wbase + (size_t)mt * 16 * C_ + k0;
      af[mt] = cat16(*(const v8bf*)(wr + 16 * hi), *(const v8bf*)(wr + 8 + 16 * hi));
    }
#pragma unroll
    for (int mt = 0; mt < 4; ++mt) {
      acc[mt][0] = WMMA_BF16(af[mt], bf0, acc[mt][0]);
      acc[mt][1] = WMMA_BF16(af[mt], bf1, acc[mt][1]);
    }

    // store prefetched tile into the other buffer; single barrier per K-step
    if (ks < C_ / 32 - 1) {
#pragma unroll
      for (int cc = 0; cc < 32; cc += 2)
        *(unsigned*)&XT[buf ^ 1][tid * XSTRIDE + cc] = stg[cc >> 1];
      __syncthreads();
    }
  }

  // store with bias; D layout: lane holds (m = r + 8*hi, n = ln)
#pragma unroll
  for (int mt = 0; mt < 4; ++mt) {
#pragma unroll
    for (int nt = 0; nt < 2; ++nt) {
      const int l = l0 + wave * 32 + nt * 16 + ln;
#pragma unroll
      for (int r = 0; r < 8; ++r) {
        const int   m   = m0 + mt * 16 + r + 8 * hi;
        const float val = acc[mt][nt][r] + bias[m];
        if (is_v) {
          vb[((size_t)b * C_ + m) * L_ + l] = (__bf16)val;
        } else if (m < 64) {
          qb[((size_t)b * L_ + l) * QK_ + m] = (__bf16)val;
        } else {
          kb[((size_t)b * L_ + l) * QK_ + (m - 64)] = (__bf16)val;
        }
      }
    }
  }
}

// ---------------------------------------------------------------------------
// Kernel 2: flash attention + residual, j stepped by 64, double-buffered P,
// one barrier per iteration, rescale-skip flag.
// 1 block = (batch b, 32 query rows). Waves 0/1 produce S/softmax/P for their
// 16-row group; all 8 waves consume both P groups against their own
// 64-channel V slice (16 PV WMMAs / iter / wave).
// ---------------------------------------------------------------------------
#define PSTRIDE 72   // bf16 elems per P row (144B: 16B aligned, conflict-free b128)

__global__ __launch_bounds__(256, 1) void attn_kernel(
    const __bf16* __restrict__ qb, const __bf16* __restrict__ kb,
    const __bf16* __restrict__ vb, const float* __restrict__ x,
    const float* __restrict__ gamma, float* __restrict__ out)
{
  __shared__ __bf16 Plds[2][2][16 * PSTRIDE];  // [group][buf][row * stride]
  __shared__ float  sc_lds[2][32];             // [buf][row]
  __shared__ int    flag_lds[2][2];            // [buf][group]: any scale != 1 ?
  __shared__ float  l_lds[32];

  const int tid  = threadIdx.x;
  const int wave = tid >> 5;
  const int lane = tid & 31;
  const int hi   = lane >> 4;
  const int ln   = lane & 15;

  const int blk = blockIdx.x;                  // B * (L/32) = 1024
  const int b   = blk >> 7;
  const int i0  = (blk & 127) << 5;

  const int  cbase = wave * 64;                // this wave's channel slice
  const bool prod  = (wave < 2);               // producer for row group `wave`

  v8f acc[2][4];                               // [row-group][16-channel tile]
#pragma unroll
  for (int g = 0; g < 2; ++g)
#pragma unroll
    for (int nt = 0; nt < 4; ++nt) acc[g][nt] = (v8f){};

  v16bf qa0 = {}, qa1 = {};
  float mrow[8], lrow[8];
#pragma unroll
  for (int r = 0; r < 8; ++r) { mrow[r] = -3.0e38f; lrow[r] = 0.f; }

  if (prod) {
    const __bf16* qrow = qb + ((size_t)b * L_ + i0 + wave * 16 + ln) * QK_;
    qa0 = cat16(*(const v8bf*)(qrow +      16 * hi), *(const v8bf*)(qrow +  8 + 16 * hi));
    qa1 = cat16(*(const v8bf*)(qrow + 32 + 16 * hi), *(const v8bf*)(qrow + 40 + 16 * hi));
  }

  const __bf16* vrow = vb + ((size_t)b * C_ + cbase + ln) * L_ + 16 * hi;

  for (int it = 0; it < L_ / 64; ++it) {
    const int j0  = it * 64;
    const int buf = it & 1;

    if (prod) {
      // S for 4 j sub-tiles: lane holds column j, k-elems = d = 32t + 16*hi + e
      v8f s[4];
#pragma unroll
      for (int u = 0; u < 4; ++u) {
        const __bf16* kr = kb + ((size_t)b * L_ + j0 + 16 * u + ln) * QK_ + 16 * hi;
        v16bf kf0 = ld_bf16x16(kr);
        v16bf kf1 = ld_bf16x16(kr + 32);
        v8f t = {};
        t = WMMA_BF16(qa0, kf0, t);
        t = WMMA_BF16(qa1, kf1, t);
        s[u] = t;
      }
      __builtin_prefetch(kb + ((size_t)b * L_ + j0 + 64 + ln) * QK_, 0, 0);

      // online softmax over 64 columns; rows r+8*hi live in this lane's half
      float sc_l[8];
      bool  ch = false;
#pragma unroll
      for (int r = 0; r < 8; ++r) {
        float t = fmaxf(fmaxf(s[0][r], s[1][r]), fmaxf(s[2][r], s[3][r]));
        t = fmaxf(t, __shfl_xor(t, 1));
        t = fmaxf(t, __shfl_xor(t, 2));
        t = fmaxf(t, __shfl_xor(t, 4));
        t = fmaxf(t, __shfl_xor(t, 8));
        const float mn = fmaxf(mrow[r], t);
        ch = ch || (mn > mrow[r]);
        const float scale = __expf(mrow[r] - mn);
        mrow[r] = mn;
        sc_l[r] = scale;
        const int m = r + 8 * hi;
        float rs = 0.f;
#pragma unroll
        for (int u = 0; u < 4; ++u) {
          const float e = __expf(s[u][r] - mn);
          rs += e;
          Plds[wave][buf][m * PSTRIDE + u * 16 + ln] = (__bf16)e;
        }
        rs += __shfl_xor(rs, 1);
        rs += __shfl_xor(rs, 2);
        rs += __shfl_xor(rs, 4);
        rs += __shfl_xor(rs, 8);
        lrow[r] = lrow[r] * scale + rs;
      }
      const int vote = __any(ch);
      if (lane == 0) {
#pragma unroll
        for (int r = 0; r < 8; ++r) sc_lds[buf][wave * 16 + r] = sc_l[r];
        flag_lds[buf][wave] = vote;
      }
      if (lane == 16) {
#pragma unroll
        for (int r = 0; r < 8; ++r) sc_lds[buf][wave * 16 + 8 + r] = sc_l[r];
      }
    }

    __syncthreads();   // P[buf] + scales visible; buf^1 free for next iteration

    // conditional accumulator rescale (skipped when running max unchanged)
    const int f0 = flag_lds[buf][0];
    const int f1 = flag_lds[buf][1];
    if (f0) {
#pragma unroll
      for (int r = 0; r < 8; ++r) {
        const float sA = sc_lds[buf][r + 8 * hi];
#pragma unroll
        for (int nt = 0; nt < 4; ++nt) acc[0][nt][r] *= sA;
      }
    }
    if (f1) {
#pragma unroll
      for (int r = 0; r < 8; ++r) {
        const float sB = sc_lds[buf][16 + r + 8 * hi];
#pragma unroll
        for (int nt = 0; nt < 4; ++nt) acc[1][nt][r] *= sB;
      }
    }

    // P A-fragments from LDS: row m = ln, elems 0..15 @ j=32h+16hi+e
    v16bf pf[2][2];
#pragma unroll
    for (int g = 0; g < 2; ++g)
#pragma unroll
      for (int h = 0; h < 2; ++h) {
        const __bf16* pr = &Plds[g][buf][ln * PSTRIDE + 32 * h + 16 * hi];
        pf[g][h] = cat16(*(const v8bf*)pr, *(const v8bf*)(pr + 8));
      }

    // shared V B-fragments: lane holds column c, k-elems = j_local = 32h + 16*hi + e
#pragma unroll
    for (int nt = 0; nt < 4; ++nt) {
#pragma unroll
      for (int h = 0; h < 2; ++h) {
        v16bf vf = ld_bf16x16(vrow + (size_t)(16 * nt) * L_ + j0 + 32 * h);
        acc[0][nt] = WMMA_BF16(pf[0][h], vf, acc[0][nt]);
        acc[1][nt] = WMMA_BF16(pf[1][h], vf, acc[1][nt]);
      }
    }
    __builtin_prefetch(vrow + j0 + 64, 0, 0);
  }

  if (prod) {
    if (lane == 0) {
#pragma unroll
      for (int r = 0; r < 8; ++r) l_lds[wave * 16 + r] = lrow[r];
    }
    if (lane == 16) {
#pragma unroll
      for (int r = 0; r < 8; ++r) l_lds[wave * 16 + 8 + r] = lrow[r];
    }
  }
  __syncthreads();

  const float g = gamma[0];
#pragma unroll
  for (int grp = 0; grp < 2; ++grp) {
#pragma unroll
    for (int r = 0; r < 8; ++r) {
      const int    m   = grp * 16 + r + 8 * hi;
      const float  inv = 1.0f / l_lds[m];
      const size_t i   = (size_t)(i0 + m);
      const size_t c0  = (size_t)b * C_ + cbase + ln;
      const size_t x0  = (c0 +  0) * L_ + i;
      const size_t x1  = (c0 + 16) * L_ + i;
      const size_t x2  = (c0 + 32) * L_ + i;
      const size_t x3  = (c0 + 48) * L_ + i;
      out[x0] = fmaf(g, acc[grp][0][r] * inv, x[x0]);
      out[x1] = fmaf(g, acc[grp][1][r] * inv, x[x1]);
      out[x2] = fmaf(g, acc[grp][2][r] * inv, x[x2]);
      out[x3] = fmaf(g, acc[grp][3][r] * inv, x[x3]);
    }
  }
}

// ---------------------------------------------------------------------------
// Host launcher
// ---------------------------------------------------------------------------
extern "C" void kernel_launch(void* const* d_in, const int* in_sizes, int n_in,
                              void* d_out, int out_size, void* d_ws, size_t ws_size,
                              hipStream_t stream) {
  const float* x     = (const float*)d_in[0];
  const float* Wq    = (const float*)d_in[1];
  const float* bq    = (const float*)d_in[2];
  const float* Wk    = (const float*)d_in[3];
  const float* bk    = (const float*)d_in[4];
  const float* Wv    = (const float*)d_in[5];
  const float* bv    = (const float*)d_in[6];
  const float* gamma = (const float*)d_in[7];
  float*       out   = (float*)d_out;

  // workspace: q/k bf16 [B][L][64], v bf16 [B][C][L], bf16 weights, stacked bias
  __bf16* qb  = (__bf16*)d_ws;
  __bf16* kb  = qb  + (size_t)B_ * L_ * QK_;
  __bf16* vb  = kb  + (size_t)B_ * L_ * QK_;
  __bf16* wqk = vb  + (size_t)B_ * C_ * L_;
  __bf16* wv  = wqk + (size_t)128 * C_;
  float*  bqk = (float*)(wv + (size_t)C_ * C_);

  prep_weights_kernel<<<(C_ * C_) / 256, 256, 0, stream>>>(Wq, bq, Wk, bk, Wv, wqk, wv, bqk);

  // q/k projection: M=128 -> grid.y=2 ; v projection: M=512 -> grid.y=8
  proj_gemm_kernel<<<dim3(L_ / 256, 2, B_), 256, 0, stream>>>(wqk, bqk, x, qb, kb, vb, 0);
  proj_gemm_kernel<<<dim3(L_ / 256, 8, B_), 256, 0, stream>>>(wv, bv, x, qb, kb, vb, 1);

  attn_kernel<<<B_ * (L_ / 32), 256, 0, stream>>>(qb, kb, vb, x, gamma, out);
}
